// CausalMultiHeadAttention_6322191860296
// MI455X (gfx1250) — matrix operand order repaired and checked by reading, not compiled, not run on hardware
//
#include <hip/hip_runtime.h>
#include <hip/hip_bf16.h>

#define D_MODEL   1024
#define NUM_HEADS 16
#define HEAD_DIM  64
#define SEQ       2048
#define BATCH     2

typedef __attribute__((ext_vector_type(16))) __bf16       v16bf;
typedef __attribute__((ext_vector_type(8)))  float        v8f;
typedef __attribute__((ext_vector_type(4)))  unsigned int u32x4;

union ABu { v16bf v; u32x4 q[2]; };
union PK8 { unsigned short s[8]; u32x4 q; };

#define WMMA_BF16(a, b, c) \
  __builtin_amdgcn_wmma_f32_16x16x32_bf16(false, (a), false, (b), (short)0, (c), false, false)

#if __has_builtin(__builtin_amdgcn_s_wait_asynccnt)
#define WAIT_ASYNC(n) __builtin_amdgcn_s_wait_asynccnt(n)
#else
#define WAIT_ASYNC(n) asm volatile("s_wait_asynccnt %0" :: "n"(n) : "memory")
#endif

__device__ __forceinline__ unsigned short f2bf(float f) {
  union { float f; unsigned u; } x; x.f = f;
  unsigned r = x.u + 0x7FFFu + ((x.u >> 16) & 1u);   // round-to-nearest-even
  return (unsigned short)(r >> 16);
}

// A operand: 16(M) x 32(K) bf16 tile. lane&15 = M row.
// lanes 0-15: K = {0..7, 16..23}; lanes 16-31: K = {8..15, 24..31}  (ISA 7.12.2)
__device__ __forceinline__ v16bf load_a(const unsigned short* base, int ld, int lane) {
  const unsigned short* p = base + (lane & 15) * ld + ((lane >> 4) << 3);
  ABu u;
  u.q[0] = *(const u32x4*)(p);
  u.q[1] = *(const u32x4*)(p + 16);
  return u.v;
}

// B operand: 32(K) x 16(N) bf16 tile, source stored column-major (row n = K-contiguous).
// lanes 0-15: K = {0..7, 16..23}; lanes 16-31: K = {8..15, 24..31}
__device__ __forceinline__ v16bf load_b(const unsigned short* base, int ld, int lane) {
  const unsigned short* p = base + (lane & 15) * ld + ((lane >> 4) << 3);
  ABu u;
  u.q[0] = *(const u32x4*)(p);
  u.q[1] = *(const u32x4*)(p + 16);
  return u.v;
}

// Async-copy a contiguous 4KB block (global -> LDS), 8 x (32 lanes x 16B).
// INST_OFFSET is applied to both LDS and global addresses (ISA async load to LDS).
__device__ __forceinline__ void async_k_chunk(unsigned lds_byte, const unsigned short* g,
                                              int lane) {
  unsigned long long ga = (unsigned long long)(uintptr_t)g + (unsigned)(lane * 16);
  unsigned la = lds_byte + (unsigned)(lane * 16);
  asm volatile("global_load_async_to_lds_b128 %0, %1, off"              :: "v"(la), "v"(ga) : "memory");
  asm volatile("global_load_async_to_lds_b128 %0, %1, off offset:512"   :: "v"(la), "v"(ga) : "memory");
  asm volatile("global_load_async_to_lds_b128 %0, %1, off offset:1024"  :: "v"(la), "v"(ga) : "memory");
  asm volatile("global_load_async_to_lds_b128 %0, %1, off offset:1536"  :: "v"(la), "v"(ga) : "memory");
  asm volatile("global_load_async_to_lds_b128 %0, %1, off offset:2048"  :: "v"(la), "v"(ga) : "memory");
  asm volatile("global_load_async_to_lds_b128 %0, %1, off offset:2560"  :: "v"(la), "v"(ga) : "memory");
  asm volatile("global_load_async_to_lds_b128 %0, %1, off offset:3072"  :: "v"(la), "v"(ga) : "memory");
  asm volatile("global_load_async_to_lds_b128 %0, %1, off offset:3584"  :: "v"(la), "v"(ga) : "memory");
}

// Async-copy V chunk: 64 rows x 64B, global row stride 4096B -> LDS row stride 64B.
__device__ __forceinline__ void async_v_chunk(unsigned lds_byte, const unsigned short* g,
                                              int lane) {
  const unsigned row = (unsigned)(lane >> 2);
  const unsigned col = (unsigned)(lane & 3) * 16u;
#pragma unroll
  for (int s = 0; s < 8; ++s) {
    unsigned long long ga = (unsigned long long)(uintptr_t)g +
                            (unsigned long long)(s * 8 + row) * 4096ull + col;
    unsigned la = lds_byte + (unsigned)(s * 8 + row) * 64u + col;
    asm volatile("global_load_async_to_lds_b128 %0, %1, off" :: "v"(la), "v"(ga) : "memory");
  }
}

// ---------------------------------------------------------------- fp32 -> bf16
__global__ void cvt_bf16_kernel(const float* __restrict__ src,
                                unsigned short* __restrict__ dst, int n) {
  int i = blockIdx.x * blockDim.x + threadIdx.x;
  int stride = gridDim.x * blockDim.x;
  for (; i < n; i += stride) dst[i] = f2bf(src[i]);
}

// ------------------------- QKV projection + RoPE, 64x64 tile per wave (fused)
// grid.x = 64 M-tiles * 16 heads, grid.y = 3 (Q/K/V), block = 1 wave.
// __launch_bounds__(32,1): single-wave kernel, take the full VGPR budget (no spills).
__global__ void __launch_bounds__(32, 1)
qkv_rope_kernel(const unsigned short* __restrict__ Xb,
                const unsigned short* __restrict__ Wq,
                const unsigned short* __restrict__ Wk,
                const unsigned short* __restrict__ Wv,
                const int* __restrict__ tp,
                unsigned short* __restrict__ Qr,
                unsigned short* __restrict__ Kr,
                unsigned short* __restrict__ Vt) {
  const int lane  = threadIdx.x;
  const int tileM = blockIdx.x & 63;
  const int h     = blockIdx.x >> 6;
  const int mat   = blockIdx.y;                    // 0=Q 1=K 2=V
  const unsigned short* W = (mat == 0) ? Wq : (mat == 1) ? Wk : Wv;
  const int g0 = tileM * 64;
  const int n0 = h * HEAD_DIM;

  const v8f vz = {0.f,0.f,0.f,0.f,0.f,0.f,0.f,0.f};
  v8f c[4][4];
#pragma unroll
  for (int i = 0; i < 4; ++i)
#pragma unroll
    for (int t = 0; t < 4; ++t) c[i][t] = vz;

  for (int k = 0; k < D_MODEL; k += 32) {
    v16bf a[4];
#pragma unroll
    for (int i = 0; i < 4; ++i)
      a[i] = load_a(Xb + (size_t)(g0 + 16 * i) * D_MODEL + k, D_MODEL, lane);
#pragma unroll
    for (int t = 0; t < 4; ++t) {
      v16bf b = load_b(W + (size_t)(n0 + t * 16) * D_MODEL + k, D_MODEL, lane);
#pragma unroll
      for (int i = 0; i < 4; ++i) c[i][t] = WMMA_BF16(a[i], b, c[i][t]);
    }
  }

  const int bIdx = g0 >> 11;            // g0 / SEQ  (64-row tiles never cross batch)
  const int s0   = g0 & (SEQ - 1);
  const int half = lane >> 4, nl = lane & 15;

  if (mat == 2) {
    // V: store transposed [b][h][dh][s] so PV's B-operand loads are contiguous.
#pragma unroll
    for (int i = 0; i < 4; ++i) {
#pragma unroll
      for (int t = 0; t < 4; ++t) {
        const int dh = t * 16 + nl;
        PK8 pk;
#pragma unroll
        for (int r = 0; r < 8; ++r) pk.s[r] = f2bf(c[i][t][r]);
        unsigned short* dst = Vt +
            ((size_t)(bIdx * NUM_HEADS + h) * HEAD_DIM + dh) * SEQ +
            s0 + 16 * i + half * 8;
        *(u32x4*)dst = pk.q;
      }
    }
  } else {
    unsigned short* Obase = ((mat == 0) ? Qr : Kr) +
        ((size_t)(bIdx * NUM_HEADS + h) * SEQ) * HEAD_DIM;
    const float qscale = (mat == 0) ? 0.125f : 1.0f;   // fold 1/sqrt(64) into Q
#pragma unroll
    for (int t = 0; t < 4; ++t) {
      const int nh = t * 16 + nl;                      // dim within head
      const float freq = __powf(10000.0f, -(float)(2 * (nh >> 1)) * (1.0f / 64.0f));
#pragma unroll
      for (int i = 0; i < 4; ++i) {
#pragma unroll
        for (int r = 0; r < 8; ++r) {
          const int s = s0 + 16 * i + half * 8 + r;
          float sn, cs;
          __sincosf((float)tp[s] * freq, &sn, &cs);
          const float val  = c[i][t][r];
          const float part = __shfl_xor(val, 1);       // rotation partner lane
          float res = (nh & 1) ? (part * sn + val * cs)
                               : (val * cs - part * sn);
          res *= qscale;
          Obase[(size_t)s * HEAD_DIM + nh] = f2bf(res);
        }
      }
    }
  }
}

// ------------------- flash attention: 32 queries/wave, async K/V double-buffer
// grid (SEQ/32, H, B), block = 1 wave, 18KB dynamic LDS.
__global__ void __launch_bounds__(32, 1)
flash_attn_kernel(const unsigned short* __restrict__ Qr,
                  const unsigned short* __restrict__ Kr,
                  const unsigned short* __restrict__ Vt,
                  unsigned short* __restrict__ Ao) {
  extern __shared__ unsigned short smem[];
  // LDS map (bytes): K[2][32][64] @0 (8KB), V[2][64][32] @8192 (8KB), P[2][16][32] @16384 (2KB)
  const int lane = threadIdx.x;
  const int q0   = blockIdx.x * 32;
  const int h    = blockIdx.y;
  const int b    = blockIdx.z;
  const int half = lane >> 4, nl = lane & 15;
  const float NINF = -__builtin_inff();

  const unsigned short* Qb = Qr + ((size_t)(b * NUM_HEADS + h) * SEQ + q0) * HEAD_DIM;
  const unsigned short* Kb = Kr + ((size_t)(b * NUM_HEADS + h) * SEQ) * HEAD_DIM;
  const unsigned short* Vb = Vt + ((size_t)(b * NUM_HEADS + h) * HEAD_DIM) * SEQ;

  v16bf aq[2][2];
#pragma unroll
  for (int i = 0; i < 2; ++i) {
    aq[i][0] = load_a(Qb + (size_t)(16 * i) * HEAD_DIM,      HEAD_DIM, lane);
    aq[i][1] = load_a(Qb + (size_t)(16 * i) * HEAD_DIM + 32, HEAD_DIM, lane);
  }

  const v8f vz = {0.f,0.f,0.f,0.f,0.f,0.f,0.f,0.f};
  v8f acc[2][4];
  float Mx[2][8], Ls[2][8];
#pragma unroll
  for (int i = 0; i < 2; ++i) {
#pragma unroll
    for (int t = 0; t < 4; ++t) acc[i][t] = vz;
#pragma unroll
    for (int r = 0; r < 8; ++r) { Mx[i][r] = NINF; Ls[i][r] = 0.f; }
  }

  // prologue: stage chunk 0 into buffer 0
  async_k_chunk(0u,    Kb, lane);                      // K rows 0..31
  async_v_chunk(8192u, Vb, lane);                      // V dims x keys 0..31

  int buf = 0;
  for (int j0 = 0; j0 <= q0; j0 += 32) {
    const bool more = (j0 + 32) <= q0;
    if (more) {                                        // prefetch next chunk
      async_k_chunk((unsigned)((buf ^ 1) * 4096),
                    Kb + (size_t)(j0 + 32) * HEAD_DIM, lane);
      async_v_chunk((unsigned)(8192 + (buf ^ 1) * 4096),
                    Vb + (size_t)(j0 + 32), lane);
    }
    if (more) { WAIT_ASYNC(16); } else { WAIT_ASYNC(0); }   // current chunk landed
    asm volatile("" ::: "memory");

    const unsigned short* Kl = smem + buf * 2048;           // [32][64]
    const unsigned short* Vl = smem + 4096 + buf * 2048;    // [64][32]
    unsigned short* Pl = smem + 8192;                       // [2][16][32]

    v16bf kb[2][2];
#pragma unroll
    for (int st = 0; st < 2; ++st) {
      kb[st][0] = load_b(Kl + st * 16 * 64,      64, lane);
      kb[st][1] = load_b(Kl + st * 16 * 64 + 32, 64, lane);
    }
    v8f sc[2][2];
#pragma unroll
    for (int i = 0; i < 2; ++i)
#pragma unroll
      for (int st = 0; st < 2; ++st) {
        v8f z = vz;
        z = WMMA_BF16(aq[i][0], kb[st][0], z);
        z = WMMA_BF16(aq[i][1], kb[st][1], z);
        sc[i][st] = z;
      }

    // mask + online softmax per 16-row block
#pragma unroll
    for (int i = 0; i < 2; ++i) {
      const int m0 = q0 + 16 * i;
#pragma unroll
      for (int st = 0; st < 2; ++st) {
        const int rel = (j0 + 16 * st) - m0;
        if (rel == 0) {                                 // diagonal tile
#pragma unroll
          for (int r = 0; r < 8; ++r) {
            const int m = r + half * 8;
            sc[i][st][r] = (nl > m) ? NINF : sc[i][st][r];
          }
        } else if (rel > 0) {                           // fully above diagonal
#pragma unroll
          for (int r = 0; r < 8; ++r) sc[i][st][r] = NINF;
        }
      }
#pragma unroll
      for (int r = 0; r < 8; ++r) {
        float t = fmaxf(sc[i][0][r], sc[i][1][r]);
        t = fmaxf(t, __shfl_xor(t, 1));
        t = fmaxf(t, __shfl_xor(t, 2));
        t = fmaxf(t, __shfl_xor(t, 4));
        t = fmaxf(t, __shfl_xor(t, 8));
        const float nm    = fmaxf(Mx[i][r], t);
        const float scale = __expf(Mx[i][r] - nm);
        Mx[i][r] = nm;
        const float p0 = __expf(sc[i][0][r] - nm);
        const float p1 = __expf(sc[i][1][r] - nm);
        float rs = p0 + p1;
        rs += __shfl_xor(rs, 1);
        rs += __shfl_xor(rs, 2);
        rs += __shfl_xor(rs, 4);
        rs += __shfl_xor(rs, 8);
        Ls[i][r] = Ls[i][r] * scale + rs;
#pragma unroll
        for (int t4 = 0; t4 < 4; ++t4) acc[i][t4][r] *= scale;
        const int m = r + half * 8;
        Pl[i * 512 + m * 32 + nl]      = f2bf(p0);
        Pl[i * 512 + m * 32 + 16 + nl] = f2bf(p1);
      }
    }
    __syncthreads();                                    // 1-wave WG: waits only
    ABu pa[2];
#pragma unroll
    for (int i = 0; i < 2; ++i) {
      const unsigned short* pp = Pl + i * 512 + nl * 32 + (half << 3);
      pa[i].q[0] = *(const u32x4*)(pp);
      pa[i].q[1] = *(const u32x4*)(pp + 16);
    }
    __syncthreads();

#pragma unroll
    for (int t4 = 0; t4 < 4; ++t4) {
      v16bf bv = load_b(Vl + t4 * 16 * 32, 32, lane);
#pragma unroll
      for (int i = 0; i < 2; ++i) acc[i][t4] = WMMA_BF16(pa[i].v, bv, acc[i][t4]);
    }
    buf ^= 1;
  }

#pragma unroll
  for (int i = 0; i < 2; ++i)
#pragma unroll
    for (int t4 = 0; t4 < 4; ++t4)
#pragma unroll
      for (int r = 0; r < 8; ++r) {
        const int m = r + half * 8;
        const float o = acc[i][t4][r] / Ls[i][r];
        Ao[(size_t)(b * SEQ + q0 + 16 * i + m) * D_MODEL +
           h * HEAD_DIM + t4 * 16 + nl] = f2bf(o);
      }
}

// ------------------------------------ output projection, 64x64 tile per wave
__global__ void __launch_bounds__(32, 1)
oproj_kernel(const unsigned short* __restrict__ Ao,
             const unsigned short* __restrict__ Wo,
             float* __restrict__ out) {
  const int lane  = threadIdx.x;
  const int tileM = blockIdx.x & 63;
  const int ng    = blockIdx.x >> 6;
  const int g0 = tileM * 64, n0 = ng * 64;

  const v8f vz = {0.f,0.f,0.f,0.f,0.f,0.f,0.f,0.f};
  v8f c[4][4];
#pragma unroll
  for (int i = 0; i < 4; ++i)
#pragma unroll
    for (int t = 0; t < 4; ++t) c[i][t] = vz;

  for (int k = 0; k < D_MODEL; k += 32) {
    v16bf a[4];
#pragma unroll
    for (int i = 0; i < 4; ++i)
      a[i] = load_a(Ao + (size_t)(g0 + 16 * i) * D_MODEL + k, D_MODEL, lane);
#pragma unroll
    for (int t = 0; t < 4; ++t) {
      v16bf b = load_b(Wo + (size_t)(n0 + t * 16) * D_MODEL + k, D_MODEL, lane);
#pragma unroll
      for (int i = 0; i < 4; ++i) c[i][t] = WMMA_BF16(a[i], b, c[i][t]);
    }
  }
  const int half = lane >> 4, nl = lane & 15;
#pragma unroll
  for (int i = 0; i < 4; ++i)
#pragma unroll
    for (int t = 0; t < 4; ++t)
#pragma unroll
      for (int r = 0; r < 8; ++r)
        out[(size_t)(g0 + 16 * i + r + half * 8) * D_MODEL +
            n0 + t * 16 + nl] = c[i][t][r];
}

// ------------------------------------------------------------------ launch
extern "C" void kernel_launch(void* const* d_in, const int* in_sizes, int n_in,
                              void* d_out, int out_size, void* d_ws, size_t ws_size,
                              hipStream_t stream) {
  (void)in_sizes; (void)n_in; (void)out_size; (void)ws_size;
  const float* x  = (const float*)d_in[0];
  const float* wq = (const float*)d_in[1];
  const float* wk = (const float*)d_in[2];
  const float* wv = (const float*)d_in[3];
  const float* wo = (const float*)d_in[4];
  const int*   tp = (const int*)d_in[5];

  char* ws = (char*)d_ws;
  const size_t XB = (size_t)BATCH * SEQ * D_MODEL * 2;      // 8 MiB
  const size_t WB = (size_t)D_MODEL * D_MODEL * 2;          // 2 MiB
  unsigned short* Xb  = (unsigned short*)(ws);
  unsigned short* Wqb = (unsigned short*)(ws + XB);
  unsigned short* Wkb = (unsigned short*)(ws + XB + WB);
  unsigned short* Wvb = (unsigned short*)(ws + XB + 2 * WB);
  unsigned short* Wob = (unsigned short*)(ws + XB + 3 * WB);
  unsigned short* Qr  = (unsigned short*)(ws + XB + 4 * WB);
  unsigned short* Kr  = (unsigned short*)(ws + 2 * XB + 4 * WB);
  unsigned short* Vt  = (unsigned short*)(ws + 3 * XB + 4 * WB);
  unsigned short* Ao  = (unsigned short*)(ws + 4 * XB + 4 * WB);

  cvt_bf16_kernel<<<dim3(1024), 256, 0, stream>>>(x,  Xb,  BATCH * SEQ * D_MODEL);
  cvt_bf16_kernel<<<dim3(512),  256, 0, stream>>>(wq, Wqb, D_MODEL * D_MODEL);
  cvt_bf16_kernel<<<dim3(512),  256, 0, stream>>>(wk, Wkb, D_MODEL * D_MODEL);
  cvt_bf16_kernel<<<dim3(512),  256, 0, stream>>>(wv, Wvb, D_MODEL * D_MODEL);
  cvt_bf16_kernel<<<dim3(512),  256, 0, stream>>>(wo, Wob, D_MODEL * D_MODEL);

  qkv_rope_kernel<<<dim3(64 * 16, 3), 32, 0, stream>>>(Xb, Wqb, Wkb, Wvb, tp,
                                                       Qr, Kr, Vt);
  flash_attn_kernel<<<dim3(SEQ / 32, NUM_HEADS, BATCH), 32, 18432, stream>>>(Qr, Kr, Vt, Ao);
  oproj_kernel<<<dim3(64 * 16), 32, 0, stream>>>(Ao, Wob, (float*)d_out);
}
